// TwoDimensionalSSM_39341900431765
// MI455X (gfx1250) — matrix operand order repaired and checked by reading, not compiled, not run on hardware
//
#include <hip/hip_runtime.h>
#include <hip/hip_bf16.h>
#include <math.h>

typedef __attribute__((ext_vector_type(16))) __bf16        v16bf;
typedef __attribute__((ext_vector_type(8)))  float         v8f;
typedef __attribute__((ext_vector_type(8)))  unsigned int  v8u;
typedef __attribute__((ext_vector_type(4)))  float         f32x4;
typedef __attribute__((ext_vector_type(4)))  unsigned int  u32x4;

#define D_CH   768
#define L_SEQ  1024
#define B_SZ   64
#define L_SIDE 32

// workspace layout (bytes)
static constexpr size_t K_OFF  = 0;                                 // D*1024 bf16 (1.5 MB)
static constexpr size_t XT_OFF = 2ull * 1024 * 1024;                // D*65536 bf16 (96 MB)
static constexpr size_t OT_OFF = XT_OFF + 768ull * 65536ull * 2ull; // D*65536 f32 (192 MB)

__device__ __forceinline__ float sigm(float v) { return 1.0f / (1.0f + __expf(-v)); }

__device__ __forceinline__ unsigned short f2bf(float f) {
  unsigned int u = __builtin_bit_cast(unsigned int, f);
  u += 0x7FFFu + ((u >> 16) & 1u);
  return (unsigned short)(u >> 16);
}

// async DMA: global -> LDS, 16B per lane, tracked by ASYNCcnt
__device__ __forceinline__ void async_g2l_b128(unsigned int lds_off, const void* gaddr) {
  asm volatile("global_load_async_to_lds_b128 %0, %1, off"
               :: "v"(lds_off), "v"((unsigned long long)(size_t)gaddr)
               : "memory");
}
// async DMA: LDS -> global, 16B per lane, tracked by ASYNCcnt
__device__ __forceinline__ void async_l2g_b128(void* gaddr, unsigned int lds_off) {
  asm volatile("global_store_async_from_lds_b128 %0, %1, off"
               :: "v"((unsigned long long)(size_t)gaddr), "v"(lds_off)
               : "memory");
}
__device__ __forceinline__ void wait_async0() {
  asm volatile("s_wait_asynccnt 0x0" ::: "memory");
}

// ---------------------------------------------------------------------------
// Kernel 1: per-channel SSM impulse response k[d, 32, 32] -> bf16
// One wave (32 lanes = j) per d. Weighted prefix scan via __shfl_up (wave32).
// ---------------------------------------------------------------------------
__global__ __launch_bounds__(32) void ssm_k(
    const float* __restrict__ A1, const float* __restrict__ A2,
    const float* __restrict__ A3, const float* __restrict__ A4,
    const float* __restrict__ B1, const float* __restrict__ B2,
    const float* __restrict__ C1, const float* __restrict__ C2,
    unsigned short* __restrict__ kout) {
  const int d = blockIdx.x;
  const int j = threadIdx.x;                 // 0..31
  const float scale = 0.7071067811865476f;   // sqrt(1/N), N=2

  float sa2[2], sa3[2], sa4[2], sb1[2], sb2[2], c1s[2], c2s[2], cp[2][5];
#pragma unroll
  for (int n = 0; n < 2; ++n) {
    const int idx = d * 2 + n;
    const float a1 = sigm(A1[idx]);
    sa2[n] = sigm(A2[idx]); sa3[n] = sigm(A3[idx]); sa4[n] = sigm(A4[idx]);
    sb1[n] = sigm(B1[idx]); sb2[n] = sigm(B2[idx]);
    c1s[n] = C1[idx] * scale; c2s[n] = C2[idx] * scale;
    const float c = 0.5f * a1;
    cp[n][0] = c;
    cp[n][1] = c * c;
    cp[n][2] = cp[n][1] * cp[n][1];
    cp[n][3] = cp[n][2] * cp[n][2];
    cp[n][4] = cp[n][3] * cp[n][3];
  }

  float hh[2] = {0.f, 0.f};
  float hv[2];
  hv[0] = (j == 0) ? sb2[0] : 0.f;
  hv[1] = (j == 0) ? sb2[1] : 0.f;

  for (int i = 0; i < L_SIDE; ++i) {
    float kij = 0.f;
#pragma unroll
    for (int n = 0; n < 2; ++n) {
      if (i > 0) hv[n] = 0.5f * (sa3[n] * hh[n] + sa4[n] * hv[n]);
      const float hvm = __shfl_up(hv[n], 1, 32);
      // u[j]: impulse at origin, else 0.5*a2*h_v[i][j-1]
      float xs = (j == 0) ? ((i == 0) ? sb1[n] : 0.f) : 0.5f * sa2[n] * hvm;
      // weighted inclusive scan: hh[j] = sum_{m<=j} (0.5 a1)^(j-m) u[m]
#pragma unroll
      for (int s = 0; s < 5; ++s) {
        const float sh = __shfl_up(xs, 1u << s, 32);
        xs += (j >= (1 << s)) ? cp[n][s] * sh : 0.f;
      }
      hh[n] = xs;
      kij += hh[n] * c1s[n] + hv[n] * c2s[n];
    }
    // boundary corrections (net: (0,0) unchanged, row0/col0 x2)
    if (i == 0 && j > 0) kij *= 2.f;
    if (j == 0 && i > 0) kij *= 2.f;
    kout[d * 1024 + i * 32 + j] = f2bf(kij);
  }
}

// ---------------------------------------------------------------------------
// Kernel 2: transpose+pack x (L,B,D) f32 -> xT bf16 in WMMA B-fragment order:
//   elem index = d*65536 + bhalf*32768 + i*1024 + btile*512 + lane*16 + e
//   lane = (jin>>4)*16 + (b&15), e = jin&15, btile = (b>>4)&1, bhalf = b>>5
// Block: 32 d-channels x (one i) x (one bhalf). Coalesced read along d;
// write phase streams LDS -> global with async B128 stores (ASYNCcnt).
// ---------------------------------------------------------------------------
__global__ __launch_bounds__(256) void xt_pack(const float* __restrict__ x,
                                               unsigned short* __restrict__ xt) {
  __shared__ unsigned short tile[32 * 1024];  // 64 KB, per-d fragment-ordered rows
  const int d0 = blockIdx.x * 32;
  const int i  = blockIdx.y;
  const int bh = blockIdx.z;
  const int t  = threadIdx.x;

#pragma unroll 4
  for (int it = 0; it < 32; ++it) {
    const int idx4 = it * 256 + t;          // one float4 of d per (jin,b')
    const int dt   = (idx4 & 7) * 4;
    const int rem  = idx4 >> 3;             // 0..1023
    const int bp   = rem & 31;              // b within half
    const int jin  = rem >> 5;              // 0..31
    const f32x4 v = *(const f32x4*)(x + (size_t)((i * 32 + jin) * 64 + bh * 32 + bp) * 768 + d0 + dt);
    const int pos = (bp >> 4) * 512 + (((jin >> 4) * 16 + (bp & 15)) * 16) + (jin & 15);
#pragma unroll
    for (int q = 0; q < 4; ++q) tile[(dt + q) * 1024 + pos] = f2bf(v[q]);
  }
  __syncthreads();

  const size_t base = (size_t)d0 * 65536 + (size_t)bh * 32768 + (size_t)i * 1024;
#pragma unroll 4
  for (int it = 0; it < 16; ++it) {
    const int w4 = it * 256 + t;            // uint4 = 8 bf16
    const int dt = w4 >> 7;
    const int wi = w4 & 127;
    async_l2g_b128(xt + base + (size_t)dt * 65536 + wi * 8,
                   (unsigned int)(size_t)&tile[dt * 1024 + wi * 8]);
  }
  wait_async0();
}

// ---------------------------------------------------------------------------
// Kernel 3: depthwise causal 2D conv as block-Toeplitz GEMM on WMMA bf16.
// Block = (d, bhalf=32 batch). 8 waves: (jtile, btile, i-parity).
// Out_i(32x32b) = sum_p T_p(32x32) @ X_{i-p}(32x32b), K=32 per wmma.
// X fragments staged with async global->LDS DMA (ASYNCcnt).
// ---------------------------------------------------------------------------
__global__ __launch_bounds__(256) void conv_wmma(
    const unsigned short* __restrict__ kbf,
    const unsigned short* __restrict__ xt,
    float* __restrict__ ot) {
  __shared__ unsigned short XF[32768];  // 64 KB: [i][btile][lane][e] fragment order
  __shared__ unsigned short KK[1024];   // 2 KB bf16 kernel for this d
  const int d  = blockIdx.x;
  const int bh = blockIdx.y;
  const int t  = threadIdx.x;

  // stage X fragments (64 KB) straight into LDS, no VGPR round-trip
  const unsigned short* xsrc = xt + (size_t)d * 65536 + (size_t)bh * 32768;
#pragma unroll 4
  for (int it = 0; it < 16; ++it) {
    const int idx = it * 256 + t;
    async_g2l_b128((unsigned int)(size_t)&XF[idx * 8], xsrc + idx * 8);
  }
  if (t < 128) {  // stage k[d] (2 KB)
    async_g2l_b128((unsigned int)(size_t)&KK[t * 8], kbf + (size_t)d * 1024 + t * 8);
  }
  wait_async0();
  __syncthreads();

  const int wave  = t >> 5, lane = t & 31;
  const int jtile = wave & 1;
  const int btile = (wave >> 1) & 1;
  const int ipar  = (wave >> 2) & 1;
  const int Mloc  = lane & 15;
  const int Khalf = lane >> 4;
  const int jo    = jtile * 16 + Mloc;

  // A-fragment gather pattern: element e of this lane is T_p[jo][K_e],
  // K_e = e + 16*Khalf; T_p[jo][ji] = k[p][jo-ji] (causal)
  int  aoff[16];
  bool aval[16];
#pragma unroll
  for (int e = 0; e < 16; ++e) {
    const int K = e + 16 * Khalf;
    const int diff = jo - K;
    aval[e] = (diff >= 0);
    aoff[e] = aval[e] ? diff : 0;
  }

  v8f acc[16] = {};
  for (int p = 0; p < 32; ++p) {
    unsigned int au[8];
#pragma unroll
    for (int m = 0; m < 8; ++m) {
      unsigned short lo = KK[p * 32 + aoff[2 * m]];
      unsigned short hi = KK[p * 32 + aoff[2 * m + 1]];
      lo = aval[2 * m] ? lo : (unsigned short)0;
      hi = aval[2 * m + 1] ? hi : (unsigned short)0;
      au[m] = (unsigned int)lo | ((unsigned int)hi << 16);
    }
    const v8u auv = {au[0], au[1], au[2], au[3], au[4], au[5], au[6], au[7]};
    const v16bf Af = __builtin_bit_cast(v16bf, auv);

#pragma unroll
    for (int tt = 0; tt < 16; ++tt) {
      const int i = ipar + 2 * tt;          // this wave's output rows
      if (i >= p) {                         // wave-uniform causal guard
        const int ip = i - p;
        const v16bf Bf = *(const v16bf*)&XF[ip * 1024 + btile * 512 + lane * 16];
        acc[tt] = __builtin_amdgcn_wmma_f32_16x16x32_bf16(
            false, Af, false, Bf, (short)0, acc[tt], false, false);
      }
    }
  }

  // epilogue: C layout -> ot[d][l=i*32+jo][b] f32
  const int bcol = bh * 32 + btile * 16 + Mloc;  // N = lane&15
#pragma unroll
  for (int tt = 0; tt < 16; ++tt) {
    const int i = ipar + 2 * tt;
#pragma unroll
    for (int r = 0; r < 8; ++r) {
      const int M = r + 8 * Khalf;
      const int l = i * 32 + jtile * 16 + M;
      ot[(size_t)d * 65536 + (size_t)l * 64 + bcol] = acc[tt][r];
    }
  }
}

// ---------------------------------------------------------------------------
// Kernel 4: un-transpose + residual + silu. Coalesced both sides via LDS tile.
// out[lb*768 + d] = silu(ot[d*65536 + lb] + x[lb*768+d]*omega[d])
// ---------------------------------------------------------------------------
__global__ __launch_bounds__(256) void fin_silu(const float* __restrict__ ot,
                                                const float* __restrict__ x,
                                                const float* __restrict__ omega,
                                                float* __restrict__ out) {
  __shared__ float tile[32][33];
  const int lb0 = blockIdx.x * 32;
  const int d0  = blockIdx.y * 32;
  const int t   = threadIdx.x;
  const int c = t & 31, r = t >> 5;
#pragma unroll
  for (int it = 0; it < 4; ++it) {
    const int row = r + it * 8;  // d index
    tile[row][c] = ot[(size_t)(d0 + row) * 65536 + lb0 + c];
  }
  __syncthreads();
#pragma unroll
  for (int it = 0; it < 4; ++it) {
    const int lbr = r + it * 8;
    const size_t g = (size_t)(lb0 + lbr) * 768 + d0 + c;
    const float v = tile[c][lbr] + x[g] * omega[d0 + c];
    out[g] = v / (1.f + __expf(-v));
  }
}

extern "C" void kernel_launch(void* const* d_in, const int* in_sizes, int n_in,
                              void* d_out, int out_size, void* d_ws, size_t ws_size,
                              hipStream_t stream) {
  (void)in_sizes; (void)n_in; (void)out_size; (void)ws_size;
  const float* x     = (const float*)d_in[0];
  const float* A1    = (const float*)d_in[1];
  const float* A2    = (const float*)d_in[2];
  const float* A3    = (const float*)d_in[3];
  const float* A4    = (const float*)d_in[4];
  const float* B1    = (const float*)d_in[5];
  const float* B2    = (const float*)d_in[6];
  const float* C1    = (const float*)d_in[7];
  const float* C2    = (const float*)d_in[8];
  const float* omega = (const float*)d_in[9];
  float* out = (float*)d_out;

  char* ws = (char*)d_ws;
  unsigned short* kbf = (unsigned short*)(ws + K_OFF);
  unsigned short* xtb = (unsigned short*)(ws + XT_OFF);
  float*          otb = (float*)(ws + OT_OFF);

  ssm_k<<<dim3(D_CH), dim3(32), 0, stream>>>(A1, A2, A3, A4, B1, B2, C1, C2, kbf);
  xt_pack<<<dim3(D_CH / 32, L_SIDE, 2), dim3(256), 0, stream>>>(x, xtb);
  conv_wmma<<<dim3(D_CH, 2), dim3(256), 0, stream>>>(kbf, xtb, otb);
  fin_silu<<<dim3(L_SEQ * B_SZ / 32, D_CH / 32), dim3(256), 0, stream>>>(otb, x, omega, out);
}
